// ArcFaceLoss_36876589204058
// MI455X (gfx1250) — matrix operand order repaired and checked by reading, not compiled, not run on hardware
//
#include <hip/hip_runtime.h>
#include <hip/hip_bf16.h>
#include <math.h>

typedef __attribute__((ext_vector_type(16))) _Float16 v16h;
typedef __attribute__((ext_vector_type(8)))  _Float16 v8h;
typedef __attribute__((ext_vector_type(8)))  float    v8f;
typedef __attribute__((ext_vector_type(4)))  unsigned int u32x4;
typedef __attribute__((ext_vector_type(8)))  int i32x8;
typedef __attribute__((ext_vector_type(4)))  int i32x4;

#define B_DIM    1024
#define D_DIM    256
#define C_DIM    50000
#define SCALE_F  30.0f
#define MARGIN_F 0.5f
#define EPS_F    1e-7f

#define WAVES     8
#define BLOCK     256
#define MROWS     128                               /* rows per block: 8 waves x 16 */
#define CHUNK_N   1024
#define NCHUNKS   ((C_DIM + CHUNK_N - 1) / CHUNK_N) /* 49 */
#define C_PAD     (NCHUNKS * CHUNK_N)               /* 50176 (zero-filled pad rows) */
#define STAGE_COLS 32
#define NSTAGES   (CHUNK_N / STAGE_COLS)            /* 32 */
#define ROW_BYTES 512                               /* D_DIM * 2 */
#define LDS_ROW   528                               /* 512 + 16B pad -> conflict-free */
#define STAGE_BYTES (STAGE_COLS * LDS_ROW)          /* 16896 */

#if __has_builtin(__builtin_amdgcn_tensor_load_to_lds) && __has_builtin(__builtin_amdgcn_s_wait_tensorcnt)
#define HAVE_TDM 1
#else
#define HAVE_TDM 0
#endif

// ---------------------------------------------------------------------------
// L2-normalize one row (D=256) per wave32 -> f16; zero-fill padded rows.
// ---------------------------------------------------------------------------
__global__ __launch_bounds__(32)
void l2norm_rows_to_f16(const float* __restrict__ in, _Float16* __restrict__ out,
                        int validRows) {
  const int row  = blockIdx.x;
  const int lane = threadIdx.x;
  _Float16* dst = out + (size_t)row * D_DIM + lane * 8;
  if (row >= validRows) {
    v8h z = {};
    *(v8h*)dst = z;
    return;
  }
  const float* p = in + (size_t)row * D_DIM + lane * 8;
  float v[8];
#pragma unroll
  for (int i = 0; i < 8; ++i) v[i] = p[i];
  float ss = 0.f;
#pragma unroll
  for (int i = 0; i < 8; ++i) ss += v[i] * v[i];
#pragma unroll
  for (int off = 16; off >= 1; off >>= 1) ss += __shfl_xor(ss, off, 32);
  const float inv = 1.0f / fmaxf(sqrtf(ss), 1e-12f);
  v8h o;
#pragma unroll
  for (int i = 0; i < 8; ++i) o[i] = (_Float16)(v[i] * inv);
  *(v8h*)dst = o;
}

// ---------------------------------------------------------------------------
// helpers
// ---------------------------------------------------------------------------
__device__ __forceinline__ unsigned lds_addr_of(const void* p) {
  return (unsigned)(size_t)(const __attribute__((address_space(3))) char*)p;
}

#if HAVE_TDM
// TDM: 1D copy of STAGE_COLS x 512B W rows into LDS, inserting 16B pad per 512B
// so LDS row stride = 528B (bank-conflict-free ds_load_b128 across 16 lanes).
__device__ __forceinline__ void tdm_load_stage(const _Float16* gsrc, unsigned ldsOff) {
  const unsigned long long ga = (unsigned long long)(size_t)gsrc;
  const unsigned elems = STAGE_COLS * D_DIM;  // 8192 elements of 2B
  u32x4 g0;
  g0[0] = 1u;                                   // count=1, user descriptor
  g0[1] = ldsOff;                               // lds_addr
  g0[2] = (unsigned)ga;                         // global_addr[31:0]
  g0[3] = (unsigned)((ga >> 32) & 0x01FFFFFFu) | (2u << 30);  // addr[56:32], type=2
  i32x8 g1;
  g1[0] = (1 << 16)                             // data_size = 2B
        | (1 << 20)                             // pad_enable
        | (6 << 22)                             // pad_interval: every 128 dwords (512B)
        | (3 << 25);                            // pad_amount: 4 dwords (16B)
  g1[1] = (int)((elems & 0xFFFFu) << 16);       // tensor_dim0[15:0]  @ bits 63:48
  g1[2] = (int)((elems >> 16) | (1u << 16));    // tensor_dim0[31:16], tensor_dim1=1
  g1[3] = (int)(elems << 16);                   // tensor_dim1 hi=0, tile_dim0 @127:112
  g1[4] = 1;                                    // tile_dim1=1, tile_dim2=0
  g1[5] = (int)elems;                           // tensor_dim0_stride low
  g1[6] = 0;
  g1[7] = 0;
  const i32x4 z4 = {0, 0, 0, 0};
  const i32x8 z8 = {0, 0, 0, 0, 0, 0, 0, 0};
  // 6-arg (clang-23 / therock) form: (g0, g1, g2, g3, extra, cpol)
  __builtin_amdgcn_tensor_load_to_lds(g0, g1, z4, z4, z8, 0);
}
#endif

// synchronous fallback staging: 256 threads copy 32 rows x 512B with 528B stride
__device__ __forceinline__ void stage_copy_sync(const _Float16* gsrc,
                                                unsigned char* buf, int tid) {
  const int row = tid >> 3;        // 0..31
  const int seg = tid & 7;         // 0..7, 64B each
  const v8h* g = (const v8h*)((const char*)gsrc + row * ROW_BYTES + seg * 64);
  v8h* d = (v8h*)(buf + row * LDS_ROW + seg * 64);
#pragma unroll
  for (int i = 0; i < 4; ++i) d[i] = g[i];
}

// one 32-column stage: 2 subtiles x (8 ds-frag loads + 8 WMMA) + branchless exp-sum
__device__ __forceinline__ void compute_stage(const v16h (&A)[8],
                                              const unsigned char* buf,
                                              int stageColBase, int l15, int half,
                                              float rs[8]) {
#pragma unroll
  for (int nsub = 0; nsub < 2; ++nsub) {
    const int col = stageColBase + nsub * 16 + l15;
    const unsigned char* rowp = buf + (nsub * 16 + l15) * LDS_ROW + half * 16;
    v16h Bf[8];
#pragma unroll
    for (int ks = 0; ks < 8; ++ks) {
      v8h lo = *(const v8h*)(rowp + ks * 64);
      v8h hi = *(const v8h*)(rowp + ks * 64 + 32);
#pragma unroll
      for (int i = 0; i < 8; ++i) { Bf[ks][i] = lo[i]; Bf[ks][i + 8] = hi[i]; }
    }
    v8f acc = {};
#pragma unroll
    for (int ks = 0; ks < 8; ++ks)
      acc = __builtin_amdgcn_wmma_f32_16x16x32_f16(false, A[ks], false, Bf[ks],
                                                   (short)0, acc, false, false);
    const bool colOK = (col < C_DIM);
#pragma unroll
    for (int r = 0; r < 8; ++r) {
      float x = fminf(fmaxf(acc[r], -1.f + EPS_F), 1.f - EPS_F);
      x = colOK ? x : -4.0f;                     // exp(30*-4-30) == 0 -> masked
      rs[r] += __expf(fmaf(SCALE_F, x, -SCALE_F));
    }
  }
}

// ---------------------------------------------------------------------------
// Fused GEMM + bounded-softmax partial sums.  grid = (B/128, NCHUNKS)
// Block: 8 waves; wave w owns rows [blk*128 + 16w, +16) with A-panel in VGPRs;
// all waves share TDM-staged 32-col W tiles in LDS (double buffered).
// ---------------------------------------------------------------------------
__global__ __launch_bounds__(BLOCK)
void arcface_gemm_fused(const _Float16* __restrict__ E,
                        const _Float16* __restrict__ W,
                        float* __restrict__ part_s) {
  __shared__ __align__(16) unsigned char stage[2][STAGE_BYTES];

  const int tid  = threadIdx.x;
  const int wave = tid >> 5;
  const int lane = tid & 31;
  const int half = lane >> 4;
  const int l15  = lane & 15;
  const int rowG      = blockIdx.x * MROWS + wave * 16;  // wave's 16-row M-tile
  const int chunkBase = blockIdx.y * CHUNK_N;

  // A-panel 16x256 f16 in 64 VGPRs (ISA 16x32 layout)
  const int arow = rowG + l15;
  v16h A[8];
#pragma unroll
  for (int ks = 0; ks < 8; ++ks) {
    const _Float16* pa = E + (size_t)arow * D_DIM + ks * 32 + half * 8;
    v8h lo = *(const v8h*)(pa);
    v8h hi = *(const v8h*)(pa + 16);
#pragma unroll
    for (int i = 0; i < 8; ++i) { A[ks][i] = lo[i]; A[ks][i + 8] = hi[i]; }
  }

  float rs[8];
#pragma unroll
  for (int r = 0; r < 8; ++r) rs[r] = 0.f;

#if HAVE_TDM
  const unsigned ldsBase = lds_addr_of(&stage[0][0]);
  if (wave == 0)
    tdm_load_stage(W + (size_t)chunkBase * D_DIM, ldsBase);
  for (int s = 0; s < NSTAGES; ++s) {
    if (wave == 0) {
      if (s + 1 < NSTAGES) {
        tdm_load_stage(W + (size_t)(chunkBase + (s + 1) * STAGE_COLS) * D_DIM,
                       ldsBase + ((s + 1) & 1) * STAGE_BYTES);
        __builtin_amdgcn_s_wait_tensorcnt(1);   // oldest (stage s) complete
      } else {
        __builtin_amdgcn_s_wait_tensorcnt(0);
      }
    }
    __syncthreads();                             // stage s visible to all waves
    compute_stage(A, &stage[s & 1][0], chunkBase + s * STAGE_COLS, l15, half, rs);
    __syncthreads();                             // buffer free for next DMA
  }
#else
  for (int s = 0; s < NSTAGES; ++s) {
    stage_copy_sync(W + (size_t)(chunkBase + s * STAGE_COLS) * D_DIM, &stage[0][0], tid);
    __syncthreads();
    compute_stage(A, &stage[0][0], chunkBase + s * STAGE_COLS, l15, half, rs);
    __syncthreads();
  }
#endif

  // sum across the 16 lanes sharing each row (xor stays within half-wave)
#pragma unroll
  for (int off = 1; off < 16; off <<= 1) {
#pragma unroll
    for (int r = 0; r < 8; ++r) rs[r] += __shfl_xor(rs[r], off, 32);
  }
  if (l15 == 0) {
#pragma unroll
    for (int r = 0; r < 8; ++r)
      part_s[(size_t)(rowG + half * 8 + r) * NCHUNKS + blockIdx.y] = rs[r];
  }
}

// ---------------------------------------------------------------------------
// Per-row label correction: dot(e_norm[row], w_norm[label]) recomputed exactly,
// emit exp-delta (modified - unmodified) and the modified label logit.
// ---------------------------------------------------------------------------
__global__ __launch_bounds__(32)
void arcface_label_fix(const _Float16* __restrict__ E, const _Float16* __restrict__ W,
                       const int* __restrict__ labels,
                       float* __restrict__ delta, float* __restrict__ lab_logit) {
  const int row = blockIdx.x, lane = threadIdx.x;
  const int l = labels[row];
  v8h e = *(const v8h*)(E + (size_t)row * D_DIM + lane * 8);
  v8h w = *(const v8h*)(W + (size_t)l * D_DIM + lane * 8);
  float d = 0.f;
#pragma unroll
  for (int i = 0; i < 8; ++i) d += (float)e[i] * (float)w[i];
#pragma unroll
  for (int off = 16; off >= 1; off >>= 1) d += __shfl_xor(d, off, 32);
  if (lane == 0) {
    const float x = fminf(fmaxf(d, -1.f + EPS_F), 1.f - EPS_F);
    const float unmod = SCALE_F * x;
    const float mod = SCALE_F * __cosf(acosf(x) + MARGIN_F);
    lab_logit[row] = mod;
    delta[row] = __expf(mod - SCALE_F) - __expf(unmod - SCALE_F);
  }
}

// ---------------------------------------------------------------------------
// Combine chunk partials (+ label delta) -> lse = 30 + log(sum) -> mean NLL.
// ---------------------------------------------------------------------------
__global__ __launch_bounds__(256)
void arcface_finalize(const float* __restrict__ part_s,
                      const float* __restrict__ delta,
                      const float* __restrict__ lab_logit,
                      float* __restrict__ out) {
  __shared__ float red[256];
  float local = 0.f;
  for (int row = threadIdx.x; row < B_DIM; row += 256) {
    float s = delta[row];
    for (int c = 0; c < NCHUNKS; ++c) s += part_s[(size_t)row * NCHUNKS + c];
    local += (SCALE_F + __logf(s)) - lab_logit[row];
  }
  red[threadIdx.x] = local;
  __syncthreads();
  for (int off = 128; off >= 1; off >>= 1) {
    if (threadIdx.x < off) red[threadIdx.x] += red[threadIdx.x + off];
    __syncthreads();
  }
  if (threadIdx.x == 0) out[0] = red[0] / (float)B_DIM;
}

// ---------------------------------------------------------------------------
extern "C" void kernel_launch(void* const* d_in, const int* in_sizes, int n_in,
                              void* d_out, int out_size, void* d_ws, size_t ws_size,
                              hipStream_t stream) {
  (void)in_sizes; (void)n_in; (void)out_size; (void)ws_size;
  const float* emb    = (const float*)d_in[0];
  const int*   labels = (const int*)d_in[1];
  const float* weight = (const float*)d_in[2];
  float*       out    = (float*)d_out;

  // workspace layout
  _Float16* e16 = (_Float16*)d_ws;                         // B*D f16
  _Float16* w16 = e16 + (size_t)B_DIM * D_DIM;             // C_PAD*D f16
  float* part_s    = (float*)(w16 + (size_t)C_PAD * D_DIM);
  float* delta     = part_s + (size_t)B_DIM * NCHUNKS;
  float* lab_logit = delta + B_DIM;

  l2norm_rows_to_f16<<<B_DIM, 32, 0, stream>>>(emb, e16, B_DIM);
  l2norm_rows_to_f16<<<C_PAD, 32, 0, stream>>>(weight, w16, C_DIM);
  arcface_gemm_fused<<<dim3(B_DIM / MROWS, NCHUNKS), BLOCK, 0, stream>>>(
      e16, w16, part_s);
  arcface_label_fix<<<B_DIM, 32, 0, stream>>>(e16, w16, labels, delta, lab_logit);
  arcface_finalize<<<1, 256, 0, stream>>>(part_s, delta, lab_logit, out);
}
